// ESN_11433202942568
// MI455X (gfx1250) — matrix operand order repaired and checked by reading, not compiled, not run on hardware
//
#include <hip/hip_runtime.h>
#include <hip/hip_bf16.h>
#include <math.h>

// Problem dims (fixed by reference)
#define B_SZ 64
#define D_SZ 128
#define T_SZ 128
#define N_SZ 2048
#define H_SZ 512
#define E_SZ 256
#define O_SZ 64

typedef __attribute__((ext_vector_type(16))) __bf16 v16bf;
typedef __attribute__((ext_vector_type(8)))  float v8f;

union BfTile { uint4 q[2]; v16bf v; };

__device__ __forceinline__ unsigned short f2bf(float f) {
  unsigned int u = __float_as_uint(f);
  u += 0x7FFFu + ((u >> 16) & 1u);           // round to nearest even
  return (unsigned short)(u >> 16);
}

// A-tile (16x32 bf16, M x K) per CDNA5 ISA 7.12.2:
// lane = hi*16 + m ; elements e<8 -> K = k0 + hi*8 + e ; e>=8 -> K = k0 + 16 + hi*8 + (e-8)
__device__ __forceinline__ v16bf load_a_tile(const unsigned short* __restrict__ base,
                                             int row_stride, int m, int hi, int k0) {
  const unsigned short* p = base + (size_t)m * row_stride + k0 + hi * 8;
  BfTile t;
  t.q[0] = *(const uint4*)(p);
  t.q[1] = *(const uint4*)(p + 16);
  return t.v;
}

// B-tile (32x16 bf16, K x N) where B[k][n] = W[n0+n][k0+k], W row-major (Nout x K):
// lane = hi*16 + n ; elements e<8 -> K = k0 + hi*8 + e ; e>=8 -> K = k0 + 16 + hi*8 + (e-8)
__device__ __forceinline__ v16bf load_b_tile(const unsigned short* __restrict__ W,
                                             int ldw, int n0, int ncol, int hi, int k0) {
  const unsigned short* p = W + (size_t)(n0 + ncol) * ldw + k0 + hi * 8;
  BfTile t;
  t.q[0] = *(const uint4*)(p);
  t.q[1] = *(const uint4*)(p + 16);
  return t.v;
}

// ---------------- prepass kernels ----------------
__global__ void cvt_bf16_kernel(const float* __restrict__ src,
                                unsigned short* __restrict__ dst, int n) {
  int i = blockIdx.x * blockDim.x + threadIdx.x;
  if (i < n) dst[i] = f2bf(src[i]);
}

// X (B, D, T) f32  ->  xbf (B, T, D) bf16
__global__ void xpose_kernel(const float* __restrict__ X,
                             unsigned short* __restrict__ xbf) {
  int i = blockIdx.x * blockDim.x + threadIdx.x;
  if (i >= B_SZ * T_SZ * D_SZ) return;
  int b = i / (T_SZ * D_SZ);
  int r = i - b * (T_SZ * D_SZ);
  int t = r / D_SZ;
  int d = r - t * D_SZ;
  xbf[i] = f2bf(X[(size_t)b * D_SZ * T_SZ + (size_t)d * T_SZ + t]);
}

// ---------------- recurrent step ----------------
// One wave computes a 16x16 tile of h_next. 512 waves total.
// acc = x_t @ W_in^T  +  h_prev @ W_res^T  (both via bf16 WMMA, f32 accum)
__global__ __launch_bounds__(128) void esn_step_kernel(
    const unsigned short* __restrict__ xbf,    // (B, T, D) bf16
    const unsigned short* __restrict__ Winb,   // (N, D)    bf16
    const unsigned short* __restrict__ Wresb,  // (N, N)    bf16
    const float* __restrict__ b_res,           // (N)
    const float* __restrict__ hprev_f,         // (B, N) f32
    const unsigned short* __restrict__ hprev_b,// (B, N) bf16
    float* __restrict__ hnext_f,
    unsigned short* __restrict__ hnext_b,
    int t) {
  const int lane = threadIdx.x & 31;
  const int wave = blockIdx.x * 4 + (threadIdx.x >> 5);
  const int mb = wave & 3;    // batch tile  (64/16)
  const int nb = wave >> 2;   // reservoir tile (2048/16)
  const int m  = lane & 15;
  const int hi = lane >> 4;

  v8f acc = {0.f, 0.f, 0.f, 0.f, 0.f, 0.f, 0.f, 0.f};

  // input projection, K over D=128
  const unsigned short* xrow = xbf + (size_t)(mb * 16) * (T_SZ * D_SZ) + (size_t)t * D_SZ;
  for (int kb = 0; kb < D_SZ; kb += 32) {
    v16bf a = load_a_tile(xrow, T_SZ * D_SZ, m, hi, kb);
    v16bf b = load_b_tile(Winb, D_SZ, nb * 16, m, hi, kb);
    acc = __builtin_amdgcn_wmma_f32_16x16x32_bf16(false, a, false, b,
                                                  (short)0, acc, false, false);
  }
  // recurrent term, K over N=2048
  const unsigned short* hrow = hprev_b + (size_t)(mb * 16) * N_SZ;
  for (int kb = 0; kb < N_SZ; kb += 32) {
    v16bf a = load_a_tile(hrow, N_SZ, m, hi, kb);
    v16bf b = load_b_tile(Wresb, N_SZ, nb * 16, m, hi, kb);
    acc = __builtin_amdgcn_wmma_f32_16x16x32_bf16(false, a, false, b,
                                                  (short)0, acc, false, false);
  }

  // epilogue: tanh + leaky integration (C/D layout: VGPR r -> M = r + hi*8, N = lane%16)
  const int n = nb * 16 + m;
  const float br = b_res[n];
#pragma unroll
  for (int r = 0; r < 8; ++r) {
    const int mm = mb * 16 + hi * 8 + r;
    const size_t idx = (size_t)mm * N_SZ + n;
    float ht = tanhf(acc[r] + br);
    float hn = 0.1f * hprev_f[idx] + 0.9f * ht;
    hnext_f[idx] = hn;
    hnext_b[idx] = f2bf(hn);
  }
}

// ---------------- LayerNorm over reservoir dim ----------------
__global__ __launch_bounds__(256) void layernorm_kernel(
    const float* __restrict__ h, const float* __restrict__ gamma,
    const float* __restrict__ beta, unsigned short* __restrict__ out_b) {
  __shared__ float s1[256];
  __shared__ float s2[256];
  const int b = blockIdx.x;
  const int tid = threadIdx.x;
  float sum = 0.f, sq = 0.f;
  for (int i = tid; i < N_SZ; i += 256) {
    float v = h[(size_t)b * N_SZ + i];
    sum += v; sq += v * v;
  }
  s1[tid] = sum; s2[tid] = sq;
  __syncthreads();
  for (int s = 128; s > 0; s >>= 1) {
    if (tid < s) { s1[tid] += s1[tid + s]; s2[tid] += s2[tid + s]; }
    __syncthreads();
  }
  const float mu  = s1[0] * (1.0f / N_SZ);
  const float var = s2[0] * (1.0f / N_SZ) - mu * mu;  // biased, matches jnp.var
  const float inv = rsqrtf(var + 1e-5f);
  for (int i = tid; i < N_SZ; i += 256) {
    float v = (h[(size_t)b * N_SZ + i] - mu) * inv * gamma[i] + beta[i];
    out_b[(size_t)b * N_SZ + i] = f2bf(v);
  }
}

// ---------------- MLP GEMM: C = act(A(64xK) @ W(NoutxK)^T + bias) ----------------
__global__ __launch_bounds__(128) void mlp_gemm_kernel(
    const unsigned short* __restrict__ Ab,   // (64, K) bf16
    const unsigned short* __restrict__ Wb,   // (Nout, K) bf16
    const float* __restrict__ bias,          // (Nout)
    float* __restrict__ outf0,               // optional f32 out (64 x Nout)
    float* __restrict__ outf1,               // optional duplicate f32 out
    unsigned short* __restrict__ outb,       // optional bf16 out
    int K, int Nout, int do_relu) {
  const int lane = threadIdx.x & 31;
  const int wave = blockIdx.x * 4 + (threadIdx.x >> 5);
  const int mb = wave & 3;
  const int nb = wave >> 2;
  const int m  = lane & 15;
  const int hi = lane >> 4;

  v8f acc = {0.f, 0.f, 0.f, 0.f, 0.f, 0.f, 0.f, 0.f};
  const unsigned short* arow = Ab + (size_t)(mb * 16) * K;
  for (int kb = 0; kb < K; kb += 32) {
    v16bf a = load_a_tile(arow, K, m, hi, kb);
    v16bf b = load_b_tile(Wb, K, nb * 16, m, hi, kb);
    acc = __builtin_amdgcn_wmma_f32_16x16x32_bf16(false, a, false, b,
                                                  (short)0, acc, false, false);
  }
  const int n = nb * 16 + m;
  const float bs = bias[n];
#pragma unroll
  for (int r = 0; r < 8; ++r) {
    const int mm = mb * 16 + hi * 8 + r;
    float v = acc[r] + bs;
    if (do_relu) v = fmaxf(v, 0.f);
    const size_t idx = (size_t)mm * Nout + n;
    if (outf0) outf0[idx] = v;
    if (outf1) outf1[idx] = v;
    if (outb)  outb[idx]  = f2bf(v);
  }
}

// ---------------- workspace layout (bytes) ----------------
static constexpr size_t SZ_WRES = (size_t)N_SZ * N_SZ * 2;       // 8 MB
static constexpr size_t SZ_WIN  = (size_t)N_SZ * D_SZ * 2;
static constexpr size_t SZ_XBF  = (size_t)B_SZ * T_SZ * D_SZ * 2;
static constexpr size_t SZ_W1   = (size_t)H_SZ * N_SZ * 2;
static constexpr size_t SZ_W2   = (size_t)E_SZ * H_SZ * 2;
static constexpr size_t SZ_W3   = (size_t)O_SZ * E_SZ * 2;
static constexpr size_t SZ_HF   = (size_t)B_SZ * N_SZ * 4;
static constexpr size_t SZ_HB   = (size_t)B_SZ * N_SZ * 2;
static constexpr size_t SZ_A1B  = (size_t)B_SZ * H_SZ * 2;
static constexpr size_t SZ_HIDB = (size_t)B_SZ * E_SZ * 2;

static constexpr size_t OFF_WRES = 0;
static constexpr size_t OFF_WIN  = OFF_WRES + SZ_WRES;
static constexpr size_t OFF_XBF  = OFF_WIN  + SZ_WIN;
static constexpr size_t OFF_W1   = OFF_XBF  + SZ_XBF;
static constexpr size_t OFF_W2   = OFF_W1   + SZ_W1;
static constexpr size_t OFF_W3   = OFF_W2   + SZ_W2;
static constexpr size_t OFF_H0F  = OFF_W3   + SZ_W3;
static constexpr size_t OFF_H1F  = OFF_H0F  + SZ_HF;
static constexpr size_t OFF_H0B  = OFF_H1F  + SZ_HF;
static constexpr size_t OFF_H1B  = OFF_H0B  + SZ_HB;
static constexpr size_t OFF_HNB  = OFF_H1B  + SZ_HB;
static constexpr size_t OFF_A1B  = OFF_HNB  + SZ_HB;
static constexpr size_t OFF_HIDB = OFF_A1B  + SZ_A1B;

extern "C" void kernel_launch(void* const* d_in, const int* in_sizes, int n_in,
                              void* d_out, int out_size, void* d_ws, size_t ws_size,
                              hipStream_t stream) {
  (void)in_sizes; (void)n_in; (void)out_size; (void)ws_size;
  const float* X     = (const float*)d_in[0];
  const float* W_in  = (const float*)d_in[1];
  const float* W_res = (const float*)d_in[2];
  const float* b_res = (const float*)d_in[3];
  const float* gamma = (const float*)d_in[4];
  const float* beta  = (const float*)d_in[5];
  const float* W1    = (const float*)d_in[6];
  const float* b1    = (const float*)d_in[7];
  const float* W2    = (const float*)d_in[8];
  const float* b2    = (const float*)d_in[9];
  const float* W3    = (const float*)d_in[10];
  const float* b3    = (const float*)d_in[11];

  char* ws = (char*)d_ws;
  unsigned short* Wresb = (unsigned short*)(ws + OFF_WRES);
  unsigned short* Winb  = (unsigned short*)(ws + OFF_WIN);
  unsigned short* xbf   = (unsigned short*)(ws + OFF_XBF);
  unsigned short* W1b   = (unsigned short*)(ws + OFF_W1);
  unsigned short* W2b   = (unsigned short*)(ws + OFF_W2);
  unsigned short* W3b   = (unsigned short*)(ws + OFF_W3);
  float*          h0f   = (float*)(ws + OFF_H0F);
  float*          h1f   = (float*)(ws + OFF_H1F);
  unsigned short* h0b   = (unsigned short*)(ws + OFF_H0B);
  unsigned short* h1b   = (unsigned short*)(ws + OFF_H1B);
  unsigned short* hnb   = (unsigned short*)(ws + OFF_HNB);
  unsigned short* a1b   = (unsigned short*)(ws + OFF_A1B);
  unsigned short* hidb  = (unsigned short*)(ws + OFF_HIDB);

  float* out = (float*)d_out;           // [output(64x64) | logits(64x64) | hidden(64x256)]
  float* out_logits = out + B_SZ * O_SZ;
  float* out_hidden = out + 2 * B_SZ * O_SZ;

  // --- prepass: bf16 conversions + X transpose ---
  auto cvt = [&](const float* s, unsigned short* d, int n) {
    cvt_bf16_kernel<<<(n + 255) / 256, 256, 0, stream>>>(s, d, n);
  };
  cvt(W_res, Wresb, N_SZ * N_SZ);
  cvt(W_in,  Winb,  N_SZ * D_SZ);
  cvt(W1,    W1b,   H_SZ * N_SZ);
  cvt(W2,    W2b,   E_SZ * H_SZ);
  cvt(W3,    W3b,   O_SZ * E_SZ);
  xpose_kernel<<<(B_SZ * T_SZ * D_SZ + 255) / 256, 256, 0, stream>>>(X, xbf);

  hipMemsetAsync(h0f, 0, SZ_HF, stream);
  hipMemsetAsync(h0b, 0, SZ_HB, stream);

  // --- sequential recurrence: T=128 step kernels, ping-pong h ---
  for (int t = 0; t < T_SZ; ++t) {
    const float*          hpf = (t & 1) ? h1f : h0f;
    const unsigned short* hpb = (t & 1) ? h1b : h0b;
    float*                hnf = (t & 1) ? h0f : h1f;
    unsigned short*       hnbf = (t & 1) ? h0b : h1b;
    esn_step_kernel<<<128, 128, 0, stream>>>(xbf, Winb, Wresb, b_res,
                                             hpf, hpb, hnf, hnbf, t);
  }
  // T even -> final state in h0f

  // --- LayerNorm + MLP head ---
  layernorm_kernel<<<B_SZ, 256, 0, stream>>>(h0f, gamma, beta, hnb);
  mlp_gemm_kernel<<<H_SZ / 16, 128, 0, stream>>>(hnb, W1b, b1,
                                                 nullptr, nullptr, a1b,
                                                 N_SZ, H_SZ, 1);
  mlp_gemm_kernel<<<E_SZ / 16, 128, 0, stream>>>(a1b, W2b, b2,
                                                 out_hidden, nullptr, hidb,
                                                 H_SZ, E_SZ, 1);
  mlp_gemm_kernel<<<O_SZ / 16, 128, 0, stream>>>(hidb, W3b, b3,
                                                 out, out_logits, nullptr,
                                                 E_SZ, O_SZ, 0);
}
